// QuantAttnBlock_44555990729125
// MI455X (gfx1250) — 16-bit and 8-bit matrix operand order repaired and checked by reading, compiled, not run on hardware
//
#include <hip/hip_runtime.h>


#define BB 4
#define CCH 256
#define HW 4096
#define NTILES 256

typedef int    v8i  __attribute__((ext_vector_type(8)));
typedef float  v8f  __attribute__((ext_vector_type(8)));
typedef __bf16 v16bf __attribute__((ext_vector_type(16)));

union BF16Frag { v16bf v; unsigned int u[8]; };

__device__ __forceinline__ unsigned short f2bf(float f) {
  unsigned u = __builtin_bit_cast(unsigned, f);
  u = (u + 0x7FFFu + ((u >> 16) & 1u)) >> 16;
  return (unsigned short)u;
}

__global__ void wcvt_kernel(const float* __restrict__ wq, const float* __restrict__ wk,
                            const float* __restrict__ wv, const float* __restrict__ wp,
                            unsigned short* __restrict__ wbf) {
  int i = blockIdx.x * 256 + threadIdx.x;
  int m = i >> 16, j = i & 65535;
  const float* src = (m == 0) ? wq : (m == 1) ? wk : (m == 2) ? wv : wp;
  wbf[m * 65536 + j] = f2bf(src[j]);
}

__global__ __launch_bounds__(256) void gn_kernel(const float* __restrict__ x,
                                                 const float* __restrict__ gamma,
                                                 const float* __restrict__ beta,
                                                 unsigned short* __restrict__ h_t) {
  int b = blockIdx.x >> 5, g = blockIdx.x & 31;
  const float* base = x + ((size_t)b * CCH + g * 8) * HW;
  __shared__ float red[512];
  float s = 0.f, ss = 0.f;
  for (int i = threadIdx.x; i < 8 * HW; i += 256) { float v = base[i]; s += v; ss += v * v; }
  red[threadIdx.x] = s; red[256 + threadIdx.x] = ss;
  __syncthreads();
  for (int off = 128; off > 0; off >>= 1) {
    if (threadIdx.x < off) {
      red[threadIdx.x]       += red[threadIdx.x + off];
      red[256 + threadIdx.x] += red[256 + threadIdx.x + off];
    }
    __syncthreads();
  }
  float mean = red[0]   * (1.0f / 32768.0f);
  float var  = red[256] * (1.0f / 32768.0f) - mean * mean;
  float rstd = rsqrtf(var + 1e-6f);
  for (int i = threadIdx.x; i < 8 * HW; i += 256) {
    int cl = i >> 12, n = i & 4095, c = g * 8 + cl;
    float v = (base[i] - mean) * rstd * gamma[c] + beta[c];
    h_t[((size_t)b * HW + n) * CCH + c] = f2bf(v);
  }
}

__global__ __launch_bounds__(256) void qkv_kernel(
    const unsigned short* __restrict__ h_t, const unsigned short* __restrict__ wbf,
    const float* __restrict__ bq, const float* __restrict__ bk, const float* __restrict__ bv,
    const float* __restrict__ dq_p, const float* __restrict__ zq_p,
    const float* __restrict__ dk_p, const float* __restrict__ zk_p,
    const float* __restrict__ dv_p, const float* __restrict__ zv_p,
    unsigned char* __restrict__ q_t, unsigned char* __restrict__ k_t,
    unsigned char* __restrict__ vmat, int* __restrict__ qsum, int* __restrict__ ksum) {
  int b = blockIdx.y, n0 = blockIdx.x * 16;
  int tid = threadIdx.x, wave = tid >> 5, lane = tid & 31;
  int nl = lane & 15, half = lane >> 4;
  __shared__ int sum_l[32];
  if (tid < 32) sum_l[tid] = 0;
  __syncthreads();
  const unsigned short* hrow = h_t + ((size_t)b * HW + n0 + nl) * CCH;
  float dqv = dq_p[0], dkv = dk_p[0], dvv = dv_p[0];
  float zqv = zq_p[0], zkv = zk_p[0], zvv = zv_p[0];

  for (int job = wave; job < 48; job += 8) {
    int mat = job / 16, o0 = (job % 16) * 16;
    const unsigned short* W = wbf + mat * 65536;
    const float* bias = (mat == 0) ? bq : (mat == 1) ? bk : bv;
    const unsigned short* wrow = W + (o0 + nl) * CCH;
    v8f acc = {};
    for (int kc = 0; kc < 8; ++kc) {
      int cb = kc * 32;
      BF16Frag A, Bf;
#pragma unroll
      for (int i = 0; i < 8; ++i) {
        int ka = cb + (i & 3) * 2 + (i >> 2) * 16 + half * 8;
        A.u[i] = *(const unsigned int*)(wrow + ka);
        int kb = cb + (i & 3) * 2 + (i >> 2) * 16 + half * 8;
        Bf.u[i] = *(const unsigned int*)(hrow + kb);
      }
      acc = __builtin_amdgcn_wmma_f32_16x16x32_bf16(false, A.v, false, Bf.v,
                                                    (short)0, acc, false, false);
    }
    float delta = (mat == 0) ? dqv : (mat == 1) ? dkv : dvv;
    float zp    = (mat == 0) ? zqv : (mat == 1) ? zkv : zvv;
    float rdelta = 1.0f / delta;
    int izp = (int)rintf(zp);
    int lsum = 0, n = n0 + nl;
#pragma unroll
    for (int r = 0; r < 8; ++r) {
      int o = o0 + r + half * 8;
      float val = acc[r] + bias[o];
      int xi = (int)rintf(val * rdelta) + izp;
      xi = xi < 0 ? 0 : (xi > 255 ? 255 : xi);
      if (mat == 0)      { q_t[((size_t)b * HW + n) * CCH + o] = (unsigned char)xi; lsum += xi; }
      else if (mat == 1) { k_t[((size_t)b * HW + n) * CCH + o] = (unsigned char)xi; lsum += xi; }
      else               { vmat[((size_t)b * CCH + o) * HW + n] = (unsigned char)xi; }
    }
    if (mat == 0)      atomicAdd(&sum_l[nl], lsum);
    else if (mat == 1) atomicAdd(&sum_l[16 + nl], lsum);
  }
  __syncthreads();
  if (tid < 16) {
    qsum[b * HW + n0 + tid] = sum_l[tid];
    ksum[b * HW + n0 + tid] = sum_l[16 + tid];
  }
}

__global__ __launch_bounds__(256) void attn_kernel(
    const float* __restrict__ x,
    const unsigned char* __restrict__ q_t, const unsigned char* __restrict__ k_t,
    const unsigned char* __restrict__ vmat,
    const int* __restrict__ qsum, const int* __restrict__ ksum,
    const unsigned short* __restrict__ wproj_bf, const float* __restrict__ bproj,
    const float* __restrict__ dq_p, const float* __restrict__ zq_p,
    const float* __restrict__ dk_p, const float* __restrict__ zk_p,
    const float* __restrict__ dv_p, const float* __restrict__ zv_p,
    const float* __restrict__ dw_p, float* __restrict__ out) {
  extern __shared__ char smem[];
  float* S = (float*)smem;
  unsigned char* A8 = (unsigned char*)(smem + 16 * HW * 4);

  int b = blockIdx.y, n0 = blockIdx.x * 16;
  int tid = threadIdx.x, wave = tid >> 5, lane = tid & 31;
  int nl = lane & 15, half = lane >> 4;

  float dq = dq_p[0], zq = zq_p[0], dk = dk_p[0], zk = zk_p[0];
  float dv = dv_p[0], zv = zv_p[0], dw = dw_p[0];
  float sc = dq * dk * 0.0625f;
  float zcorr = 256.0f * zq * zk;

  v8i aq[4];
  const unsigned char* qrow = q_t + ((size_t)b * HW + n0 + nl) * CCH;
#pragma unroll
  for (int kc = 0; kc < 4; ++kc)
#pragma unroll
    for (int i = 0; i < 8; ++i) {
      int ko = kc * 64 + (i & 1) * 4 + ((i >> 1) & 1) * 16 + (i >> 2) * 32 + half * 8;
      aq[kc][i] = *(const int*)(qrow + ko);
    }
  float qs[8];
#pragma unroll
  for (int r = 0; r < 8; ++r) qs[r] = (float)qsum[b * HW + n0 + r + half * 8];

  for (int mc = 0; mc < 32; ++mc) {
    int m0 = wave * 512 + mc * 16;
    const unsigned char* krow = k_t + ((size_t)b * HW + m0 + nl) * CCH;
    v8i acc = {};
#pragma unroll
    for (int kc = 0; kc < 4; ++kc) {
      v8i bfrag;
#pragma unroll
      for (int i = 0; i < 8; ++i) {
        int ko = kc * 64 + (i & 1) * 4 + ((i >> 1) & 1) * 16 + (i >> 2) * 32 + half * 8;
        bfrag[i] = *(const int*)(krow + ko);
      }
      acc = __builtin_amdgcn_wmma_i32_16x16x64_iu8(false, aq[kc], false, bfrag,
                                                   acc, false, false);
    }
    float ks = (float)ksum[b * HW + m0 + nl];
#pragma unroll
    for (int r = 0; r < 8; ++r)
      S[(r + half * 8) * HW + m0 + nl] = sc * ((float)acc[r] - zk * qs[r] - zq * ks + zcorr);
  }
  __syncthreads();

  int row = tid >> 4, sub = tid & 15;
  float* scrF = (float*)A8;
  float mx = -3.4e38f;
  for (int m = sub; m < HW; m += 16) mx = fmaxf(mx, S[row * HW + m]);
  scrF[row * 16 + sub] = mx;
  __syncthreads();
  float rmax = scrF[row * 16];
#pragma unroll
  for (int i = 1; i < 16; ++i) rmax = fmaxf(rmax, scrF[row * 16 + i]);
  __syncthreads();
  float ps = 0.f;
  for (int m = sub; m < HW; m += 16) {
    float e = __expf(S[row * HW + m] - rmax);
    S[row * HW + m] = e; ps += e;
  }
  scrF[row * 16 + sub] = ps;
  __syncthreads();
  float rsum = 0.f;
#pragma unroll
  for (int i = 0; i < 16; ++i) rsum += scrF[row * 16 + i];
  __syncthreads();

  float qscale = (1.0f / rsum) / dw;
  int myasum = 0;
  for (int m = sub; m < HW; m += 16) {
    int xi = (int)rintf(S[row * HW + m] * qscale);
    xi = xi < 0 ? 0 : (xi > 255 ? 255 : xi);
    A8[row * HW + m] = (unsigned char)xi;
    myasum += xi;
  }
  __syncthreads();
  int* Sint = (int*)S;
  Sint[row * 16 + sub] = myasum;
  __syncthreads();
  if (sub == 0) {
    int t = 0;
    for (int i = 0; i < 16; ++i) t += Sint[row * 16 + i];
    Sint[256 + row] = t;
  }
  __syncthreads();

  unsigned short* hstage = (unsigned short*)(smem + 2048);
  float asum_n = (float)Sint[256 + nl];
  float dvw = dv * dw;
  for (int hc = 0; hc < 2; ++hc) {
    int c0 = (wave + hc * 8) * 16;
    const unsigned char* vrow = vmat + ((size_t)b * CCH + c0 + nl) * HW;
    v8i acc = {};
    for (int mb = 0; mb < 64; ++mb) {
      int mbase = mb * 64;
      v8i av, bp;
#pragma unroll
      for (int i = 0; i < 8; ++i) {
        int ma = mbase + (i & 1) * 4 + ((i >> 1) & 1) * 16 + (i >> 2) * 32 + half * 8;
        av[i] = *(const int*)(vrow + ma);
        int mo = mbase + (i & 1) * 4 + ((i >> 1) & 1) * 16 + (i >> 2) * 32 + half * 8;
        bp[i] = *(const int*)(A8 + nl * HW + mo);
      }
      acc = __builtin_amdgcn_wmma_i32_16x16x64_iu8(false, av, false, bp,
                                                   acc, false, false);
    }
#pragma unroll
    for (int r = 0; r < 8; ++r) {
      int c = c0 + r + half * 8;
      hstage[c * 16 + nl] = f2bf(dvw * ((float)acc[r] - zv * asum_n));
    }
  }
  __syncthreads();

  for (int ho = 0; ho < 2; ++ho) {
    int o0 = (wave + ho * 8) * 16;
    const unsigned short* wrow = wproj_bf + (o0 + nl) * CCH;
    v8f acc = {};
#pragma unroll
    for (int kc = 0; kc < 8; ++kc) {
      int cb = kc * 32;
      BF16Frag A, Bf;
#pragma unroll
      for (int i = 0; i < 8; ++i) {
        int ka = cb + (i & 3) * 2 + (i >> 2) * 16 + half * 8;
        A.u[i] = *(const unsigned int*)(wrow + ka);
        int kb = cb + (i & 3) * 2 + (i >> 2) * 16 + half * 8;
        unsigned lo = hstage[kb * 16 + nl];
        unsigned hi = hstage[(kb + 1) * 16 + nl];
        Bf.u[i] = lo | (hi << 16);
      }
      acc = __builtin_amdgcn_wmma_f32_16x16x32_bf16(false, A.v, false, Bf.v,
                                                    (short)0, acc, false, false);
    }
#pragma unroll
    for (int r = 0; r < 8; ++r) {
      int o = o0 + r + half * 8;
      size_t idx = ((size_t)b * CCH + o) * HW + n0 + nl;
      out[idx] = x[idx] + acc[r] + bproj[o];
    }
  }
}

extern "C" void kernel_launch(void* const* d_in, const int* in_sizes, int n_in,
                              void* d_out, int out_size, void* d_ws, size_t ws_size,
                              hipStream_t stream) {
  const float* x    = (const float*)d_in[0];
  const float* gns  = (const float*)d_in[1];
  const float* gnb  = (const float*)d_in[2];
  const float* wq   = (const float*)d_in[3];
  const float* bq   = (const float*)d_in[4];
  const float* wk   = (const float*)d_in[5];
  const float* bk   = (const float*)d_in[6];
  const float* wv   = (const float*)d_in[7];
  const float* bv   = (const float*)d_in[8];
  const float* wp   = (const float*)d_in[9];
  const float* bp   = (const float*)d_in[10];
  const float* dq   = (const float*)d_in[11];
  const float* zq   = (const float*)d_in[12];
  const float* dk   = (const float*)d_in[13];
  const float* zk   = (const float*)d_in[14];
  const float* dv   = (const float*)d_in[15];
  const float* zv   = (const float*)d_in[16];
  const float* dw   = (const float*)d_in[17];
  float* out        = (float*)d_out;

  char* ws = (char*)d_ws;
  unsigned short* wbf  = (unsigned short*)(ws);
  unsigned short* h_t  = (unsigned short*)(ws + 524288);
  unsigned char*  q_t  = (unsigned char*)(ws + 524288 + 8388608);
  unsigned char*  k_t  = q_t + (size_t)BB * HW * CCH;
  unsigned char*  vm   = k_t + (size_t)BB * HW * CCH;
  int* qsum            = (int*)(vm + (size_t)BB * HW * CCH);
  int* ksum            = qsum + BB * HW;

  wcvt_kernel<<<1024, 256, 0, stream>>>(wq, wk, wv, wp, wbf);
  gn_kernel<<<BB * 32, 256, 0, stream>>>(x, gns, gnb, h_t);
  qkv_kernel<<<dim3(NTILES, BB), 256, 0, stream>>>(
      h_t, wbf, bq, bk, bv, dq, zq, dk, zk, dv, zv, q_t, k_t, vm, qsum, ksum);

  const unsigned short* wproj_bf = wbf + 3 * 65536;
  size_t smem = (size_t)16 * HW * 4 + (size_t)16 * HW;
  hipFuncSetAttribute((const void*)attn_kernel,
                      hipFuncAttributeMaxDynamicSharedMemorySize, (int)smem);
  attn_kernel<<<dim3(NTILES, BB), 256, smem, stream>>>(
      x, q_t, k_t, vm, qsum, ksum, wproj_bf, bp,
      dq, zq, dk, zk, dv, zv, dw, out);
}
